// InteractionNetwork_62663572848869
// MI455X (gfx1250) — matrix operand order repaired and checked by reading, not compiled, not run on hardware
//
#include <hip/hip_runtime.h>
#include <hip/hip_bf16.h>

typedef __attribute__((ext_vector_type(16))) _Float16 v16h;
typedef __attribute__((ext_vector_type(8)))  float    v8f;
typedef __attribute__((ext_vector_type(4)))  float    f32x4;

constexpr int B  = 4;
constexpr int N  = 2048;
constexpr int E  = 8192;
constexpr int OD = 3;
constexpr int RD = 4;
constexpr int ED = 4;
constexpr int H  = 8;
constexpr int BE   = B * E;        // 32768 edge rows
constexpr int M_IN = 2 * OD + RD;  // 10
constexpr int C_IN = OD + ED;      // 7
constexpr int LSTR = 20;           // padded LDS row stride (floats): rows M and M+8 land in disjoint bank blocks

// ---------------------------------------------------------------- helpers ---

__device__ __forceinline__ float eluf(float x) {
  return x > 0.f ? x : (__expf(x) - 1.f);
}

__device__ __forceinline__ v8f wmma16(v16h a, v16h b, v8f c) {
  // D(f32 16x16) = A(f16 16x32) * B(f16 32x16) + C
  return __builtin_amdgcn_wmma_f32_16x16x32_f16(false, a, false, b, (short)0, c,
                                                false, false);
}

// A-frag 16x32 f16. lane: M = lane&15.
// lo lanes: halves 0..7 -> K=0..7, halves 8..15 -> K=16..23; hi lanes: +8.
template <int Kin>
__device__ __forceinline__ v16h make_afrag(const float* f, bool hi) {
  v16h a = {};
#pragma unroll
  for (int j = 0; j < 16; ++j) {
    const int klo = (j & 7) + ((j >= 8) ? 16 : 0);
    const int khi = klo + 8;
    const float lo = (klo < Kin) ? f[klo] : 0.f;
    const float hv = (khi < Kin) ? f[khi] : 0.f;
    a[j] = (_Float16)(hi ? hv : lo);
  }
  return a;
}

// B-frag 32x16 f16 from row-major W[Kin][Nout]. lane: Ncol = lane&15.
// lo lanes: halves 0..7 -> K=0..7, halves 8..15 -> K=16..23; hi lanes: +8. Pads are zero.
template <int Kin, int Nout>
__device__ __forceinline__ v16h load_bfrag(const float* __restrict__ W, int lane) {
  v16h b = {};
  const int  n  = lane & 15;
  const bool hi = lane >= 16;
  if (n < Nout) {
#pragma unroll
    for (int j = 0; j < 16; ++j) {
      const int   klo = (j & 7) + ((j >= 8) ? 16 : 0);
      const float lo = (klo < Kin)       ? W[klo * Nout + n]       : 0.f;
      const float hv = ((klo + 8) < Kin) ? W[(klo + 8) * Nout + n] : 0.f;
      b[j] = (_Float16)(hi ? hv : lo);
    }
  }
  return b;
}

// one dense layer on a 16-row tile: D = act(A*W + bias)
template <int Kin, int Nout, bool ACT>
__device__ __forceinline__ v8f mlp_layer(const float* fin,
                                         const float* __restrict__ W,
                                         const float* __restrict__ bias,
                                         int lane) {
  const bool hi = lane >= 16;
  v16h a = make_afrag<Kin>(fin, hi);
  v16h b = load_bfrag<Kin, Nout>(W, lane);
  v8f  c = {};
  v8f  d = wmma16(a, b, c);
  const int   n  = lane & 15;
  const float bn = (n < Nout) ? bias[n] : 0.f;
#pragma unroll
  for (int r = 0; r < 8; ++r) {
    const float v = d[r] + bn;
    d[r] = ACT ? eluf(v) : v;
  }
  return d;
}

// D-frag (N in lanes, M in VGPRs) -> LDS tile [M][n] so next layer can read
// rows as its A K-dimension. Per-wave private region; LDS is in-order per wave.
__device__ __forceinline__ void xpose_store(float* scr, v8f d, int lane) {
  const int n  = lane & 15;
  const int mo = (lane >= 16) ? 8 : 0;
#pragma unroll
  for (int r = 0; r < 8; ++r)
    scr[(r + mo) * LSTR + n] = d[r];
  asm volatile("s_wait_dscnt 0" ::: "memory");
}

// ---------------------------------------------------------------- kernels ---

// Stage 1: recover indices from one-hot rows. One wave per edge row, b128 NT
// streaming loads. This stage carries ~537 MB and dominates runtime.
__global__ __launch_bounds__(256) void onehot_idx(const float* __restrict__ Ro,
                                                  const float* __restrict__ Ri,
                                                  int* __restrict__ roi,
                                                  int* __restrict__ rii) {
  const int wid  = (int)((blockIdx.x * blockDim.x + threadIdx.x) >> 5);
  const int lane = threadIdx.x & 31;
  const f32x4* r0 = (const f32x4*)(Ro + (size_t)wid * N);
  const f32x4* r1 = (const f32x4*)(Ri + (size_t)wid * N);
  int i0 = 0, i1 = 0;
#pragma unroll 4
  for (int it = 0; it < N / 128; ++it) {
    const int v4 = it * 32 + lane;
    const f32x4 a = __builtin_nontemporal_load(r0 + v4);
    const f32x4 c = __builtin_nontemporal_load(r1 + v4);
    const int nb = v4 * 4;
    if (a.x > 0.5f) i0 = nb;     if (a.y > 0.5f) i0 = nb + 1;
    if (a.z > 0.5f) i0 = nb + 2; if (a.w > 0.5f) i0 = nb + 3;
    if (c.x > 0.5f) i1 = nb;     if (c.y > 0.5f) i1 = nb + 1;
    if (c.z > 0.5f) i1 = nb + 2; if (c.w > 0.5f) i1 = nb + 3;
  }
#pragma unroll
  for (int off = 16; off > 0; off >>= 1) {
    const int t0 = __shfl_xor(i0, off, 32);
    const int t1 = __shfl_xor(i1, off, 32);
    i0 = i0 > t0 ? i0 : t0;
    i1 = i1 > t1 ? i1 : t1;
  }
  if (lane == 0) { roi[wid] = i0; rii[wid] = i1; }
}

__global__ __launch_bounds__(256) void zero_f32(float* __restrict__ p, int n) {
  const int i = blockIdx.x * blockDim.x + threadIdx.x;
  if (i < n) p[i] = 0.f;
}

// Stage 2: phi_R1 over edges (16 edges per wave) + scatter-add Eff into nodes.
__global__ __launch_bounds__(256) void edge_phiR1(
    const float* __restrict__ X,  const float* __restrict__ Ra,
    const float* __restrict__ W1, const float* __restrict__ b1,
    const float* __restrict__ W2, const float* __restrict__ b2,
    const float* __restrict__ W3, const float* __restrict__ b3,
    const float* __restrict__ W4, const float* __restrict__ b4,
    const int* __restrict__ roi,  const int* __restrict__ rii,
    float* __restrict__ Eff, float* __restrict__ Anod) {
  __shared__ float scr[8][16 * LSTR];
  const int lane = threadIdx.x & 31;
  const int wv   = threadIdx.x >> 5;
  const int tile = blockIdx.x * 8 + wv;
  const int b    = tile / (E / 16);
  const int e0   = (tile % (E / 16)) * 16;
  const int m    = lane & 15;
  const size_t eg = (size_t)b * E + e0 + m;

  float feat[M_IN];
  {
    const int ro = roi[eg];
    const int ri = rii[eg];
    const float* Xb = X + (size_t)b * OD * N;
    feat[0] = Xb[0 * N + ro]; feat[1] = Xb[1 * N + ro]; feat[2] = Xb[2 * N + ro];
    feat[3] = Xb[0 * N + ri]; feat[4] = Xb[1 * N + ri]; feat[5] = Xb[2 * N + ri];
    const f32x4 ra = *(const f32x4*)(Ra + eg * RD);
    feat[6] = ra.x; feat[7] = ra.y; feat[8] = ra.z; feat[9] = ra.w;
  }

  float* sw = scr[wv];
  v8f d = mlp_layer<M_IN, H, true>(feat, W1, b1, lane);
  xpose_store(sw, d, lane);
  d = mlp_layer<H, H, true>(sw + m * LSTR, W2, b2, lane);
  xpose_store(sw, d, lane);
  d = mlp_layer<H, H, true>(sw + m * LSTR, W3, b3, lane);
  xpose_store(sw, d, lane);
  d = mlp_layer<H, ED, false>(sw + m * LSTR, W4, b4, lane);

  const int n  = lane & 15;
  const int mo = (lane >= 16) ? 8 : 0;
  if (n < ED) {
#pragma unroll
    for (int r = 0; r < 8; ++r) {
      const size_t eg2 = (size_t)b * E + e0 + r + mo;
      const float  v   = d[r];
      Eff[eg2 * ED + n] = v;
      const int dst = rii[eg2];
      atomicAdd(&Anod[((size_t)b * N + dst) * ED + n], v);
    }
  }
}

// Stage 3: phi_O over nodes (16 nodes per wave).
__global__ __launch_bounds__(256) void node_phiO(
    const float* __restrict__ X,  const float* __restrict__ Anod,
    const float* __restrict__ W1, const float* __restrict__ b1,
    const float* __restrict__ W2, const float* __restrict__ b2,
    const float* __restrict__ W3, const float* __restrict__ b3,
    float* __restrict__ Xt2) {
  __shared__ float scr[8][16 * LSTR];
  const int lane = threadIdx.x & 31;
  const int wv   = threadIdx.x >> 5;
  const int tile = blockIdx.x * 8 + wv;
  const int b    = tile / (N / 16);
  const int n0   = (tile % (N / 16)) * 16;
  const int m    = lane & 15;
  const int node = n0 + m;

  float feat[C_IN];
  {
    const float* Xb = X + (size_t)b * OD * N;
    feat[0] = Xb[0 * N + node]; feat[1] = Xb[1 * N + node]; feat[2] = Xb[2 * N + node];
    const f32x4 av = *(const f32x4*)(Anod + ((size_t)b * N + node) * ED);
    feat[3] = av.x; feat[4] = av.y; feat[5] = av.z; feat[6] = av.w;
  }

  float* sw = scr[wv];
  v8f d = mlp_layer<C_IN, H, true>(feat, W1, b1, lane);
  xpose_store(sw, d, lane);
  d = mlp_layer<H, H, true>(sw + m * LSTR, W2, b2, lane);
  xpose_store(sw, d, lane);
  d = mlp_layer<H, 3, false>(sw + m * LSTR, W3, b3, lane);

  const int n  = lane & 15;
  const int mo = (lane >= 16) ? 8 : 0;
  if (n < 3) {
#pragma unroll
    for (int r = 0; r < 8; ++r)
      Xt2[((size_t)b * N + n0 + r + mo) * 3 + n] = d[r];
  }
}

// Stage 4: phi_R2 over edges -> sigmoid -> output.
__global__ __launch_bounds__(256) void edge_phiR2(
    const float* __restrict__ Xt2, const float* __restrict__ Eff,
    const float* __restrict__ W1, const float* __restrict__ b1,
    const float* __restrict__ W2, const float* __restrict__ b2,
    const float* __restrict__ W3, const float* __restrict__ b3,
    const float* __restrict__ W4, const float* __restrict__ b4,
    const int* __restrict__ roi,  const int* __restrict__ rii,
    float* __restrict__ out) {
  __shared__ float scr[8][16 * LSTR];
  const int lane = threadIdx.x & 31;
  const int wv   = threadIdx.x >> 5;
  const int tile = blockIdx.x * 8 + wv;
  const int b    = tile / (E / 16);
  const int e0   = (tile % (E / 16)) * 16;
  const int m    = lane & 15;
  const size_t eg = (size_t)b * E + e0 + m;

  float feat[M_IN];
  {
    const int ro = roi[eg];
    const int ri = rii[eg];
    const float* Tb = Xt2 + (size_t)b * N * 3;
    feat[0] = Tb[ri * 3 + 0]; feat[1] = Tb[ri * 3 + 1]; feat[2] = Tb[ri * 3 + 2];
    feat[3] = Tb[ro * 3 + 0]; feat[4] = Tb[ro * 3 + 1]; feat[5] = Tb[ro * 3 + 2];
    const f32x4 ev = *(const f32x4*)(Eff + eg * ED);
    feat[6] = ev.x; feat[7] = ev.y; feat[8] = ev.z; feat[9] = ev.w;
  }

  float* sw = scr[wv];
  v8f d = mlp_layer<M_IN, H, true>(feat, W1, b1, lane);
  xpose_store(sw, d, lane);
  d = mlp_layer<H, H, true>(sw + m * LSTR, W2, b2, lane);
  xpose_store(sw, d, lane);
  d = mlp_layer<H, H, true>(sw + m * LSTR, W3, b3, lane);
  xpose_store(sw, d, lane);
  d = mlp_layer<H, 1, false>(sw + m * LSTR, W4, b4, lane);

  const int n  = lane & 15;
  const int mo = (lane >= 16) ? 8 : 0;
  if (n == 0) {
#pragma unroll
    for (int r = 0; r < 8; ++r)
      out[(size_t)b * E + e0 + r + mo] = 1.f / (1.f + __expf(-d[r]));
  }
}

// ------------------------------------------------------------------ launch ---

extern "C" void kernel_launch(void* const* d_in, const int* in_sizes, int n_in,
                              void* d_out, int out_size, void* d_ws, size_t ws_size,
                              hipStream_t stream) {
  const float* X    = (const float*)d_in[0];
  const float* Ra   = (const float*)d_in[1];
  const float* Ro   = (const float*)d_in[2];
  const float* Ri   = (const float*)d_in[3];
  const float* r1W1 = (const float*)d_in[4];  const float* r1b1 = (const float*)d_in[5];
  const float* r1W2 = (const float*)d_in[6];  const float* r1b2 = (const float*)d_in[7];
  const float* r1W3 = (const float*)d_in[8];  const float* r1b3 = (const float*)d_in[9];
  const float* r1W4 = (const float*)d_in[10]; const float* r1b4 = (const float*)d_in[11];
  const float* r2W1 = (const float*)d_in[12]; const float* r2b1 = (const float*)d_in[13];
  const float* r2W2 = (const float*)d_in[14]; const float* r2b2 = (const float*)d_in[15];
  const float* r2W3 = (const float*)d_in[16]; const float* r2b3 = (const float*)d_in[17];
  const float* r2W4 = (const float*)d_in[18]; const float* r2b4 = (const float*)d_in[19];
  const float* oW1  = (const float*)d_in[20]; const float* ob1  = (const float*)d_in[21];
  const float* oW2  = (const float*)d_in[22]; const float* ob2  = (const float*)d_in[23];
  const float* oW3  = (const float*)d_in[24]; const float* ob3  = (const float*)d_in[25];

  // workspace layout (all 16B-aligned slices)
  int*   roi  = (int*)d_ws;                       // BE
  int*   rii  = roi + BE;                         // BE
  float* Eff  = (float*)(rii + BE);               // BE * ED
  float* Anod = Eff + (size_t)BE * ED;            // B * N * ED
  float* Xt2  = Anod + (size_t)B * N * ED;        // B * N * 3
  float* out  = (float*)d_out;                    // BE

  onehot_idx<<<BE / 8, 256, 0, stream>>>(Ro, Ri, roi, rii);
  zero_f32<<<(B * N * ED) / 256, 256, 0, stream>>>(Anod, B * N * ED);
  edge_phiR1<<<(BE / 16) / 8, 256, 0, stream>>>(X, Ra,
      r1W1, r1b1, r1W2, r1b2, r1W3, r1b3, r1W4, r1b4, roi, rii, Eff, Anod);
  node_phiO<<<((B * N) / 16) / 8, 256, 0, stream>>>(X, Anod,
      oW1, ob1, oW2, ob2, oW3, ob3, Xt2);
  edge_phiR2<<<(BE / 16) / 8, 256, 0, stream>>>(Xt2, Eff,
      r2W1, r2b1, r2W2, r2b2, r2W3, r2b3, r2W4, r2b4, roi, rii, out);
}
